// GraphSAGE_85332410237297
// MI455X (gfx1250) — matrix operand order repaired and checked by reading, not compiled, not run on hardware
//
#include <hip/hip_runtime.h>
#include <hip/hip_bf16.h>
#include <math.h>

typedef __attribute__((ext_vector_type(16))) _Float16 v16h;
typedef __attribute__((ext_vector_type(8)))  _Float16 v8h;
typedef __attribute__((ext_vector_type(8)))  float    v8f;

#define N_NODES_C   50000
#define N_EDGES_C   1600000
#define DIM_C       128
#define DIM_OUT_C   40
#define DIM_PAD_C   48
#define N_GRAPHS_C  64

__device__ __forceinline__ void atomic_fadd(float* p, float v) {
  (void)__hip_atomic_fetch_add(p, v, __ATOMIC_RELAXED, __HIP_MEMORY_SCOPE_AGENT);
}

// monotonic float->uint key: preserves ordering under unsigned max
__device__ __forceinline__ unsigned f32key(float f) {
  unsigned b = __float_as_uint(f);
  return (b & 0x80000000u) ? ~b : (b | 0x80000000u);
}
__device__ __forceinline__ float keyf32(unsigned k) {
  unsigned b = (k & 0x80000000u) ? (k & 0x7FFFFFFFu) : ~k;
  return __uint_as_float(b);
}

__global__ void k_zero_f32(float* __restrict__ p, int n) {
  int i = blockIdx.x * blockDim.x + threadIdx.x;
  if (i < n) p[i] = 0.0f;
}

__global__ void k_fill_u32(unsigned* __restrict__ p, unsigned v, int n) {
  int i = blockIdx.x * blockDim.x + threadIdx.x;
  if (i < n) p[i] = v;
}

// One wave per edge: 32 lanes x float4 = 128-dim row; L2-resident f32 atomics.
__global__ void k_scatter(const float* __restrict__ feat, const int* __restrict__ src,
                          const int* __restrict__ dst, float* __restrict__ agg,
                          float* __restrict__ deg, int nEdges) {
  const int lane = threadIdx.x & 31;
  const int e = blockIdx.x * (blockDim.x >> 5) + (threadIdx.x >> 5);
  if (e >= nEdges) return;
  const int s = src[e];
  const int d = dst[e];
  const float4 v = ((const float4*)(feat + (size_t)s * DIM_C))[lane];
  float* ad = agg + (size_t)d * DIM_C + lane * 4;
  atomic_fadd(ad + 0, v.x);
  atomic_fadd(ad + 1, v.y);
  atomic_fadd(ad + 2, v.z);
  atomic_fadd(ad + 3, v.w);
  if (deg != nullptr && lane == 0) atomic_fadd(deg + d, 1.0f);
}

// f32 -> f16 convert; if deg != null, normalize row by max(deg,1) first (dim fixed 128)
__global__ void k_cvt_f16(const float* __restrict__ in, const float* __restrict__ deg,
                          _Float16* __restrict__ out, int count) {
  int i = blockIdx.x * blockDim.x + threadIdx.x;
  if (i >= count) return;
  float s = 1.0f;
  if (deg != nullptr) s = 1.0f / fmaxf(deg[i >> 7], 1.0f);
  out[i] = (_Float16)(in[i] * s);
}

// W f32 [K][Nw] row-major -> Wt f16 [Npad][K] (transposed, zero-padded columns)
__global__ void k_prep_wt(const float* __restrict__ W, _Float16* __restrict__ Wt,
                          int K, int Nw, int Npad) {
  int i = blockIdx.x * blockDim.x + threadIdx.x;
  if (i >= Npad * K) return;
  int n = i / K;
  int k = i - n * K;
  Wt[i] = (n < Nw) ? (_Float16)W[(size_t)k * Nw + n] : (_Float16)0.0f;
}

__global__ void k_prep_bias(const float* __restrict__ b, float* __restrict__ bp,
                            int Nw, int Npad) {
  int i = threadIdx.x;
  if (i < Npad) bp[i] = (i < Nw) ? b[i] : 0.0f;
}

// out[16x16 tile] = A1 @ B1^T + A2 @ B2^T + bias  (+optional ReLU)
// A: row-major f16 [rows][K]; Bt: f16 [Ncols][K] (pre-transposed weights).
// One wave per 16x16 output tile; fragments match ISA 7.12.2 layouts:
//   A lane=row, two contiguous 8-half loads (K = hi*8.. and 16+hi*8..)
//   B lane=col, two contiguous 8-half loads (K = hi*8.. and 16+hi*8..)
//   D row = hi*8 + v, col = lane&15
__global__ void k_sage_gemm_wmma(const _Float16* __restrict__ A1, const _Float16* __restrict__ B1t,
                                 const _Float16* __restrict__ A2, const _Float16* __restrict__ B2t,
                                 const float* __restrict__ bias, float* __restrict__ out,
                                 int rowTiles, int colTiles, int K, int ldo, int doRelu) {
  const int lane = threadIdx.x & 31;
  const int tile = blockIdx.x * (blockDim.x >> 5) + (threadIdx.x >> 5);
  if (tile >= rowTiles * colTiles) return;  // wave-uniform: EXEC stays all-ones for WMMA
  const int tr = tile / colTiles;
  const int tc = tile - tr * colTiles;
  const int m  = lane & 15;
  const int hi = lane >> 4;

  const _Float16* a1 = A1 + (size_t)(tr * 16 + m) * K;
  const _Float16* a2 = A2 + (size_t)(tr * 16 + m) * K;
  const _Float16* b1 = B1t + (size_t)(tc * 16 + m) * K;
  const _Float16* b2 = B2t + (size_t)(tc * 16 + m) * K;

  v8f acc = {};
  union Frag { v16h v; v8h h[2]; };
  for (int k0 = 0; k0 < K; k0 += 32) {
    Frag fa1, fa2, fb1, fb2;
    fa1.h[0] = *(const v8h*)(a1 + k0 + hi * 8);
    fa1.h[1] = *(const v8h*)(a1 + k0 + 16 + hi * 8);
    fa2.h[0] = *(const v8h*)(a2 + k0 + hi * 8);
    fa2.h[1] = *(const v8h*)(a2 + k0 + 16 + hi * 8);
    fb1.h[0] = *(const v8h*)(b1 + k0 + hi * 8);
    fb1.h[1] = *(const v8h*)(b1 + k0 + 16 + hi * 8);
    fb2.h[0] = *(const v8h*)(b2 + k0 + hi * 8);
    fb2.h[1] = *(const v8h*)(b2 + k0 + 16 + hi * 8);
    acc = __builtin_amdgcn_wmma_f32_16x16x32_f16(false, fa1.v, false, fb1.v,
                                                 (short)0, acc, false, false);
    acc = __builtin_amdgcn_wmma_f32_16x16x32_f16(false, fa2.v, false, fb2.v,
                                                 (short)0, acc, false, false);
  }

  const int colBase = tc * 16;
  const float bv = bias[colBase + m];
  float* orow = out + (size_t)(tr * 16 + hi * 8) * ldo + colBase + m;
#pragma unroll
  for (int r = 0; r < 8; ++r) {
    float val = acc[r] + bv;
    if (doRelu) val = fmaxf(val, 0.0f);
    orow[(size_t)r * ldo] = val;
  }
}

// per node: log_softmax over first 40 cols of padded logits; segment-max pool via uint atomicMax
__global__ void k_logsoftmax_pool(const float* __restrict__ logits, const int* __restrict__ batch,
                                  float* __restrict__ outLS, unsigned* __restrict__ pooled,
                                  int nNodes) {
  int n = blockIdx.x * blockDim.x + threadIdx.x;
  if (n >= nNodes) return;
  const float* row = logits + (size_t)n * DIM_PAD_C;
  float mx = -INFINITY;
  for (int c = 0; c < DIM_OUT_C; ++c) mx = fmaxf(mx, row[c]);
  float ssum = 0.0f;
  for (int c = 0; c < DIM_OUT_C; ++c) ssum += expf(row[c] - mx);
  const float lse = mx + logf(ssum);
  const int g = batch[n];
  unsigned* prow = pooled + (size_t)g * DIM_OUT_C;
  float* o = outLS + (size_t)n * DIM_OUT_C;
  for (int c = 0; c < DIM_OUT_C; ++c) {
    float v = row[c];
    atomicMax(prow + c, f32key(v));
    o[c] = v - lse;
  }
}

__global__ void k_mean_pool(const unsigned* __restrict__ pooled, float* __restrict__ outH) {
  int c = threadIdx.x;
  if (c >= DIM_OUT_C) return;
  float s = 0.0f;
  for (int g = 0; g < N_GRAPHS_C; ++g) s += keyf32(pooled[(size_t)g * DIM_OUT_C + c]);
  outH[c] = s * (1.0f / N_GRAPHS_C);
}

extern "C" void kernel_launch(void* const* d_in, const int* in_sizes, int n_in,
                              void* d_out, int out_size, void* d_ws, size_t ws_size,
                              hipStream_t stream) {
  (void)in_sizes; (void)n_in; (void)out_size; (void)ws_size;

  const float* x    = (const float*)d_in[0];
  const int*   ei   = (const int*)d_in[1];
  const int*   bat  = (const int*)d_in[2];
  const float* W1l  = (const float*)d_in[3];
  const float* b1   = (const float*)d_in[4];
  const float* W1r  = (const float*)d_in[5];
  const float* W2l  = (const float*)d_in[6];
  const float* b2   = (const float*)d_in[7];
  const float* W2r  = (const float*)d_in[8];
  const int* src = ei;
  const int* dst = ei + N_EDGES_C;
  float* out = (float*)d_out;

  // --- workspace carve (256B aligned) ---
  char* ws = (char*)d_ws;
  size_t off = 0;
  auto carve = [&](size_t bytes) -> char* {
    char* p = ws + off;
    off += (bytes + 255) & ~(size_t)255;
    return p;
  };
  float*    degF    = (float*)carve((size_t)N_NODES_C * 4);
  float*    aggF    = (float*)carve((size_t)N_NODES_C * DIM_C * 4);
  _Float16* aggH    = (_Float16*)carve((size_t)N_NODES_C * DIM_C * 2);
  _Float16* xh      = (_Float16*)carve((size_t)N_NODES_C * DIM_C * 2);
  float*    h1F     = (float*)carve((size_t)N_NODES_C * DIM_C * 4);
  _Float16* h1h     = (_Float16*)carve((size_t)N_NODES_C * DIM_C * 2);
  float*    logitsP = (float*)carve((size_t)N_NODES_C * DIM_PAD_C * 4);
  _Float16* W1lT    = (_Float16*)carve((size_t)DIM_C * DIM_C * 2);
  _Float16* W1rT    = (_Float16*)carve((size_t)DIM_C * DIM_C * 2);
  _Float16* W2lT    = (_Float16*)carve((size_t)DIM_PAD_C * DIM_C * 2);
  _Float16* W2rT    = (_Float16*)carve((size_t)DIM_PAD_C * DIM_C * 2);
  float*    b2p     = (float*)carve((size_t)DIM_PAD_C * 4);
  unsigned* pooled  = (unsigned*)carve((size_t)N_GRAPHS_C * DIM_OUT_C * 4);

  const int BT = 256;           // 8 wave32s per block
  const int WPB = BT / 32;
  const int cnt = N_NODES_C * DIM_C;

  // ---- layer 1 ----
  k_zero_f32<<<(N_NODES_C + BT - 1) / BT, BT, 0, stream>>>(degF, N_NODES_C);
  k_zero_f32<<<(cnt + BT - 1) / BT, BT, 0, stream>>>(aggF, cnt);

  k_prep_wt<<<(DIM_C * DIM_C + BT - 1) / BT, BT, 0, stream>>>(W1l, W1lT, DIM_C, DIM_C, DIM_C);
  k_prep_wt<<<(DIM_C * DIM_C + BT - 1) / BT, BT, 0, stream>>>(W1r, W1rT, DIM_C, DIM_C, DIM_C);
  k_prep_wt<<<(DIM_PAD_C * DIM_C + BT - 1) / BT, BT, 0, stream>>>(W2l, W2lT, DIM_C, DIM_OUT_C, DIM_PAD_C);
  k_prep_wt<<<(DIM_PAD_C * DIM_C + BT - 1) / BT, BT, 0, stream>>>(W2r, W2rT, DIM_C, DIM_OUT_C, DIM_PAD_C);
  k_prep_bias<<<1, 64, 0, stream>>>(b2, b2p, DIM_OUT_C, DIM_PAD_C);

  k_scatter<<<(N_EDGES_C + WPB - 1) / WPB, BT, 0, stream>>>(x, src, dst, aggF, degF, N_EDGES_C);

  k_cvt_f16<<<(cnt + BT - 1) / BT, BT, 0, stream>>>(aggF, degF, aggH, cnt);
  k_cvt_f16<<<(cnt + BT - 1) / BT, BT, 0, stream>>>(x, nullptr, xh, cnt);

  const int tiles1 = (N_NODES_C / 16) * (DIM_C / 16);
  k_sage_gemm_wmma<<<(tiles1 + WPB - 1) / WPB, BT, 0, stream>>>(
      aggH, W1lT, xh, W1rT, b1, h1F, N_NODES_C / 16, DIM_C / 16, DIM_C, DIM_C, 1);

  // ---- layer 2 ----
  k_zero_f32<<<(cnt + BT - 1) / BT, BT, 0, stream>>>(aggF, cnt);
  k_scatter<<<(N_EDGES_C + WPB - 1) / WPB, BT, 0, stream>>>(h1F, src, dst, aggF, nullptr, N_EDGES_C);

  k_cvt_f16<<<(cnt + BT - 1) / BT, BT, 0, stream>>>(aggF, degF, aggH, cnt);
  k_cvt_f16<<<(cnt + BT - 1) / BT, BT, 0, stream>>>(h1F, nullptr, h1h, cnt);

  const int tiles2 = (N_NODES_C / 16) * (DIM_PAD_C / 16);
  k_sage_gemm_wmma<<<(tiles2 + WPB - 1) / WPB, BT, 0, stream>>>(
      aggH, W2lT, h1h, W2rT, b2p, logitsP, N_NODES_C / 16, DIM_PAD_C / 16, DIM_C, DIM_PAD_C, 0);

  // ---- pooling + log_softmax ----
  k_fill_u32<<<(N_GRAPHS_C * DIM_OUT_C + BT - 1) / BT, BT, 0, stream>>>(
      pooled, 0x007FFFFFu /* key(-inf) */, N_GRAPHS_C * DIM_OUT_C);
  k_logsoftmax_pool<<<(N_NODES_C + BT - 1) / BT, BT, 0, stream>>>(
      logitsP, bat, out + DIM_OUT_C, pooled, N_NODES_C);
  k_mean_pool<<<1, 64, 0, stream>>>(pooled, out);
}
